// LocallyEnhancedConvTransformer_76647986364710
// MI455X (gfx1250) — matrix operand order repaired and checked by reading, not compiled, not run on hardware
//
#include <hip/hip_runtime.h>
#include <hip/hip_bf16.h>

// ---------------------------------------------------------------------------
// MI455X (gfx1250, wave32) implementation of LocallyEnhancedConvTransformer.
// GEMMs: v_wmma_f32_16x16x32_bf16 with LDS-staged, double-buffered tiles.
//   A tiles -> GLOBAL_LOAD_ASYNC_TO_LDS_B128 (ASYNCcnt)
//   B tiles -> TENSOR_LOAD_TO_LDS via TDM descriptor (TENSORcnt)
// Attention upsample+softmax+AV fused in LDS (411 MB tensor never built).
// ---------------------------------------------------------------------------

typedef __attribute__((ext_vector_type(16))) __bf16 v16bf;
typedef __attribute__((ext_vector_type(8)))  __bf16 v8bf;
typedef __attribute__((ext_vector_type(8)))  float  v8f;
typedef __attribute__((ext_vector_type(4)))  unsigned int u32x4;
typedef __attribute__((ext_vector_type(8)))  int  i32x8;
typedef __attribute__((ext_vector_type(4)))  int  i32x4;

#if __has_builtin(__builtin_amdgcn_global_load_async_to_lds_b128)
#define HAVE_ASYNC_LDS 1
#else
#define HAVE_ASYNC_LDS 0
#endif
#if __has_builtin(__builtin_amdgcn_tensor_load_to_lds)
#define HAVE_TDM 1
#else
#define HAVE_TDM 0
#endif

#define DIMC  256
#define HEADS 8
#define RESR  56
#define HIDC  1024
#define HWTOK 3136
#define BATCH 16

#define GF_GELU    1
#define GF_SCATTER 2
#define DW_INGELU  1
#define DW_OUTGELU 2

__device__ __forceinline__ float gelu_f(float x) {
    return 0.5f * x * (1.0f + erff(x * 0.70710678118654752f));
}

// 16-byte global -> LDS copy (async if available)
__device__ __forceinline__ void cp16(const __bf16* g, __bf16* l) {
#if HAVE_ASYNC_LDS
    __builtin_amdgcn_global_load_async_to_lds_b128((i32x4*)g, (i32x4*)l, 0, 0);
#else
    *(v8bf*)l = *(const v8bf*)g;
#endif
}

__device__ __forceinline__ void async_wait0() {
#if HAVE_ASYNC_LDS
#if __has_builtin(__builtin_amdgcn_s_wait_asynccnt)
    __builtin_amdgcn_s_wait_asynccnt(0);
#else
    asm volatile("s_wait_asynccnt 0x0" ::: "memory");
#endif
#endif
}

__device__ __forceinline__ void tensor_wait0() {
#if HAVE_TDM
#if __has_builtin(__builtin_amdgcn_s_wait_tensorcnt)
    __builtin_amdgcn_s_wait_tensorcnt(0);
#else
    asm volatile("s_wait_tensorcnt 0x0" ::: "memory");
#endif
#endif
}

#if HAVE_TDM
// TDM: load a 64-row x 32-col bf16 tile (row stride K elements) into LDS.
// D# group0: count=1 | lds_addr | global_addr | type=2
// D# group1: data_size=2B; tensor_dim0=32, tensor_dim1=64; tile 32x64; stride=K
__device__ __forceinline__ void tdm_load_B_tile(const __bf16* gsrc, unsigned ldsOff, int K) {
    unsigned long long ga = (unsigned long long)(size_t)gsrc;
    u32x4 g0 = { 1u, ldsOff, (unsigned)ga,
                 (unsigned)((ga >> 32) & 0x01ffffffull) | (2u << 30) };
    i32x8 g1 = { 0x10000, 32 << 16, 64 << 16, 32 << 16, 64, K, 0, 0 };
    i32x4 gz = { 0, 0, 0, 0 };
    i32x8 gz8 = { 0, 0, 0, 0, 0, 0, 0, 0 };
    __builtin_amdgcn_tensor_load_to_lds(g0, g1, gz, gz, gz8, 0);
}
#endif

__device__ __forceinline__ v16bf frag_from_lds(const __bf16* p, int stride2) {
    v8bf lo = *(const v8bf*)p;
    v8bf hi = *(const v8bf*)(p + stride2);
    v16bf f;
#pragma unroll
    for (int i = 0; i < 8; ++i) { f[i] = lo[i]; f[i + 8] = hi[i]; }
    return f;
}

// ---------------------------------------------------------------------------
// WMMA GEMM: D[M,N] = A[M,K](bf16 row major) * Bw([N][K] bf16, weight layout)
// Block = 256 threads (8 waves). Tile 128(M) x 64(N), K-step 32.
// A: double-buffered LDS via async loads; B: double-buffered LDS via TDM.
// All 5 fragments of a k-step are loaded before the 4 WMMAs issue, so the
// LDS latency is covered by one dscnt wait and the WMMAs go back-to-back.
// ---------------------------------------------------------------------------
__global__ __launch_bounds__(256) void gemm_bf16_wmma(
    const __bf16* __restrict__ A, const __bf16* __restrict__ Bw,
    int M, int N, int K,
    const float* __restrict__ bias, const float* __restrict__ scale,
    const float* __restrict__ shift, const float* __restrict__ resid,
    float* __restrict__ outF, int flags)
{
    __shared__ __align__(16) __bf16 sA[2][128 * 32];   // 2 x 8 KB
    __shared__ __align__(16) __bf16 sB[2][64 * 32];    // 2 x 4 KB

    int tid  = threadIdx.x;
    int lane = tid & 31;
    int wave = tid >> 5;
    int m0blk = blockIdx.y * 128;
    int n0    = blockIdx.x * 64;
    int hf = lane >> 4;
    int lr = lane & 15;

    // stage helpers --------------------------------------------------------
    auto prefetch_A = [&](int k0, int buf) {
#pragma unroll
        for (int h = 0; h < 2; ++h) {
            int c = tid + h * 256;          // 512 x 16B = 8 KB
            int row = c >> 2, q = c & 3;
            cp16(A + (size_t)(m0blk + row) * K + k0 + q * 8,
                 &sA[buf][row * 32 + q * 8]);
        }
    };
    auto stage_B = [&](int k0, int buf) {
#if HAVE_TDM
        if (tid < 32)
            tdm_load_B_tile(Bw + (size_t)n0 * K + k0,
                            (unsigned)(size_t)(void*)&sB[buf][0], K);
#else
        int row = tid >> 2, q = tid & 3;    // 256 x 16B = 4 KB
        cp16(Bw + (size_t)(n0 + row) * K + k0 + q * 8,
             &sB[buf][row * 32 + q * 8]);
#endif
    };

    v8f acc[4];
#pragma unroll
    for (int j = 0; j < 4; ++j)
#pragma unroll
        for (int i = 0; i < 8; ++i) acc[j][i] = 0.0f;

    int nK = K >> 5;
    stage_B(0, 0);
    prefetch_A(0, 0);

    for (int kk = 0; kk < nK; ++kk) {
        async_wait0();
        tensor_wait0();
        __syncthreads();
        int cur = kk & 1;
        if (kk + 1 < nK) {
            prefetch_A((kk + 1) * 32, cur ^ 1);
            stage_B((kk + 1) * 32, cur ^ 1);
        }
        // Load ALL fragments for this k-step first (one dscnt wait), then
        // issue the 4 WMMAs back-to-back (XDL co-execution friendly).
        // A fragment: lane lr = row, K chunk {hf*16..+15} (contiguous 32B)
        v16bf af = frag_from_lds(&sA[cur][(wave * 16 + lr) * 32 + hf * 16], 8);
        // B fragments: lane lr = col, K chunk {hf*16..+15} (contiguous 32B)
        v16bf bf0 = frag_from_lds(&sB[cur][(0 * 16 + lr) * 32 + hf * 16], 8);
        v16bf bf1 = frag_from_lds(&sB[cur][(1 * 16 + lr) * 32 + hf * 16], 8);
        v16bf bf2 = frag_from_lds(&sB[cur][(2 * 16 + lr) * 32 + hf * 16], 8);
        v16bf bf3 = frag_from_lds(&sB[cur][(3 * 16 + lr) * 32 + hf * 16], 8);
        acc[0] = __builtin_amdgcn_wmma_f32_16x16x32_bf16(
            false, af, false, bf0, (short)0, acc[0], false, false);
        acc[1] = __builtin_amdgcn_wmma_f32_16x16x32_bf16(
            false, af, false, bf1, (short)0, acc[1], false, false);
        acc[2] = __builtin_amdgcn_wmma_f32_16x16x32_bf16(
            false, af, false, bf2, (short)0, acc[2], false, false);
        acc[3] = __builtin_amdgcn_wmma_f32_16x16x32_bf16(
            false, af, false, bf3, (short)0, acc[3], false, false);
    }

    int m0 = m0blk + wave * 16;
#pragma unroll
    for (int j = 0; j < 4; ++j) {
        int n = n0 + j * 16 + lr;
        float sc = scale ? scale[n] : 1.0f;
        float sh = shift ? shift[n] : 0.0f;
        float bi = bias ? bias[n] : 0.0f;
#pragma unroll
        for (int i = 0; i < 8; ++i) {
            int r = m0 + hf * 8 + i;
            float t = acc[j][i] + bi;
            if (scale) t = t * sc + sh;
            if (flags & GF_GELU) t = gelu_f(t);
            if (resid) t += resid[(size_t)r * N + n];
            if (flags & GF_SCATTER) {
                int bb = r / HWTOK, pp = r - bb * HWTOK;
                outF[((size_t)bb * N + n) * HWTOK + pp] = t;
            } else {
                outF[(size_t)r * N + n] = t;
            }
        }
    }
}

// ---------------------------------------------------------------------------
// LayerNorm over C=256, emit bf16 GEMM A-matrix. One block per token row.
// ---------------------------------------------------------------------------
__global__ __launch_bounds__(256) void ln_to_bf16(
    const float* __restrict__ x, const float* __restrict__ g,
    const float* __restrict__ b, __bf16* __restrict__ out)
{
    __shared__ float red[256];
    int r = blockIdx.x, t = threadIdx.x;
    float v = x[(size_t)r * DIMC + t];
    red[t] = v; __syncthreads();
    for (int s = 128; s > 0; s >>= 1) { if (t < s) red[t] += red[t + s]; __syncthreads(); }
    float mean = red[0] * (1.0f / DIMC);
    __syncthreads();
    float d = v - mean;
    red[t] = d * d; __syncthreads();
    for (int s = 128; s > 0; s >>= 1) { if (t < s) red[t] += red[t + s]; __syncthreads(); }
    float rstd = rsqrtf(red[0] * (1.0f / DIMC) + 1e-6f);
    out[(size_t)r * DIMC + t] = (__bf16)(d * rstd * g[t] + b[t]);
}

__global__ void cvt_bf16(const float* __restrict__ in, __bf16* __restrict__ out, int n) {
    int i = blockIdx.x * 256 + threadIdx.x;
    if (i < n) out[i] = (__bf16)in[i];
}

__global__ void bn_coef(const float* __restrict__ g, const float* __restrict__ b,
                        const float* __restrict__ m, const float* __restrict__ v,
                        float* __restrict__ sc, float* __restrict__ sh, int n) {
    int i = blockIdx.x * 256 + threadIdx.x;
    if (i < n) { float s = g[i] * rsqrtf(v[i] + 1e-5f); sc[i] = s; sh[i] = b[i] - m[i] * s; }
}

__global__ __launch_bounds__(256) void tok2nchw(const float* __restrict__ in,
                                                float* __restrict__ out) {
    int r = blockIdx.x, c = threadIdx.x;
    int b = r / HWTOK, p = r - b * HWTOK;
    out[((size_t)b * DIMC + c) * HWTOK + p] = in[(size_t)r * DIMC + c];
}

// ---------------------------------------------------------------------------
// Depthwise 3x3 conv (SAME, zero pad) + optional bias + BN + optional GELUs.
// grid = (56, B*C), block = 64 (x). outTok != nullptr scatters bf16 token-
// major [b*HW+p][C] (GEMM A for f3); else f32 NCHW.
// ---------------------------------------------------------------------------
__global__ __launch_bounds__(64) void dw3x3_bn(
    const float* __restrict__ in, const float* __restrict__ w,
    const float* __restrict__ bias, const float* __restrict__ sc,
    const float* __restrict__ sh, float* __restrict__ outF,
    __bf16* __restrict__ outTok, int C, int flags)
{
    int x = threadIdx.x; if (x >= RESR) return;
    int y = blockIdx.x;
    int bc = blockIdx.y;
    int c = bc % C;
    const float* base = in + (size_t)bc * HWTOK;
    const float* wc = w + c * 9;
    float acc = 0.0f;
#pragma unroll
    for (int dy = -1; dy <= 1; ++dy) {
        int yy = y + dy;
        if (yy < 0 || yy >= RESR) continue;
#pragma unroll
        for (int dx = -1; dx <= 1; ++dx) {
            int xx = x + dx;
            if (xx < 0 || xx >= RESR) continue;
            float vv = base[yy * RESR + xx];
            if (flags & DW_INGELU) vv = gelu_f(vv);
            acc += vv * wc[(dy + 1) * 3 + (dx + 1)];
        }
    }
    if (bias) acc += bias[c];
    acc = acc * sc[c] + sh[c];
    if (flags & DW_OUTGELU) acc = gelu_f(acc);
    int p = y * RESR + x;
    if (outTok) {
        int b = bc / C;
        outTok[((size_t)b * HWTOK + p) * C + c] = (__bf16)acc;
    } else {
        outF[(size_t)bc * HWTOK + p] = acc;
    }
}

// 7x7 mean pool 56x56 -> 8x8; grid = B*C, block = 64
__global__ __launch_bounds__(64) void pool7(const float* __restrict__ in,
                                            float* __restrict__ out) {
    int bc = blockIdx.x, t = threadIdx.x;
    int qy = t >> 3, qx = t & 7;
    const float* base = in + (size_t)bc * HWTOK + (qy * 7) * RESR + qx * 7;
    float s = 0.0f;
#pragma unroll
    for (int i = 0; i < 7; ++i)
#pragma unroll
        for (int j = 0; j < 7; ++j) s += base[i * RESR + j];
    out[(size_t)bc * 64 + t] = s * (1.0f / 49.0f);
}

// attn0[b,h,n,m] = sum_d q[b,h*32+d,n] * k[b,h*32+d,m]; grid (64, B*H)
__global__ __launch_bounds__(64) void qk_attn(const float* __restrict__ q,
                                              const float* __restrict__ k,
                                              float* __restrict__ attn0) {
    int m = threadIdx.x, n = blockIdx.x, bh = blockIdx.y;
    int b = bh >> 3, h = bh & 7;
    const float* qp = q + ((size_t)b * DIMC + h * 32) * 64 + n;
    const float* kp = k + ((size_t)b * DIMC + h * 32) * 64 + m;
    float s = 0.0f;
#pragma unroll
    for (int d = 0; d < 32; ++d) s += qp[(size_t)d * 64] * kp[(size_t)d * 64];
    attn0[((size_t)bh * 64 + n) * 64 + m] = s;
}

// bilinear index helper, upscale factor 7, align_corners=False
__device__ __forceinline__ void upidx7(int i, int n, int& i0, int& i1, float& t) {
    float src = fmaxf((i + 0.5f) * (1.0f / 7.0f) - 0.5f, 0.0f);
    int a = (int)src;              // floor, src >= 0
    if (a > n - 1) a = n - 1;
    i0 = a;
    i1 = (a + 1 < n) ? a + 1 : n - 1;
    t = src - (float)a;
}

// ---------------------------------------------------------------------------
// Fused: bilinear upsample (8x8 -> 56x56) of attention logits, * C^-0.5,
// softmax over 3136, then p @ V. One block per (b,h,query n). Probabilities
// live in LDS; the 411 MB upsampled tensor is never written. Row max over
// the upsampled field equals row max over the 64 raw logits (grid points
// are attained exactly at i = 7k+3).
// ---------------------------------------------------------------------------
__global__ __launch_bounds__(256) void attn_softmax_av(
    const float* __restrict__ attn0, const float* __restrict__ vtok,
    float* __restrict__ y4)
{
    __shared__ float sa[64];
    __shared__ float sp[HWTOK];
    __shared__ float red[256];
    __shared__ float pd[8][32];
    int n = blockIdx.x, bh = blockIdx.y;
    int b = bh >> 3, h = bh & 7;
    int t = threadIdx.x;
    if (t < 64) sa[t] = attn0[((size_t)bh * 64 + n) * 64 + t] * 0.0625f; // 256^-0.5
    __syncthreads();
    red[t] = (t < 64) ? sa[t] : -3.0e38f;
    __syncthreads();
    for (int s = 32; s > 0; s >>= 1) { if (t < s) red[t] = fmaxf(red[t], red[t + s]); __syncthreads(); }
    float mx = red[0];
    __syncthreads();
    float zp = 0.0f;
    for (int m = t; m < HWTOK; m += 256) {
        int my = m / RESR, mxp = m - my * RESR;
        int y0, y1, x0, x1; float ty, tx;
        upidx7(my, 8, y0, y1, ty);
        upidx7(mxp, 8, x0, x1, tx);
        float v00 = sa[y0 * 8 + x0], v01 = sa[y0 * 8 + x1];
        float v10 = sa[y1 * 8 + x0], v11 = sa[y1 * 8 + x1];
        float val = (v00 * (1.f - tx) + v01 * tx) * (1.f - ty)
                  + (v10 * (1.f - tx) + v11 * tx) * ty;
        float e = __expf(val - mx);
        sp[m] = e;
        zp += e;
    }
    red[t] = zp; __syncthreads();
    for (int s = 128; s > 0; s >>= 1) { if (t < s) red[t] += red[t + s]; __syncthreads(); }
    float invZ = 1.0f / red[0];
    __syncthreads();
    int d = t & 31, ch = t >> 5;
    const float* vp = vtok + (size_t)b * HWTOK * DIMC + h * 32 + d;
    float s = 0.0f;
    int mBeg = ch * 392, mEnd = mBeg + 392;  // 8*392 == 3136
    for (int m = mBeg; m < mEnd; ++m) s += sp[m] * vp[(size_t)m * DIMC];
    pd[ch][d] = s;
    __syncthreads();
    if (t < 32) {
        float tot = 0.0f;
#pragma unroll
        for (int c2 = 0; c2 < 8; ++c2) tot += pd[c2][t];
        y4[((size_t)b * DIMC + h * 32 + t) * 64 + n] = tot * invZ;  // NCHW 8x8
    }
}

// bilinear upsample 8x8 -> 56x56 per (b,c); grid (56, B*C), block 64
__global__ __launch_bounds__(64) void up7x(const float* __restrict__ in,
                                           float* __restrict__ out) {
    int x = threadIdx.x; if (x >= RESR) return;
    int y = blockIdx.x, bc = blockIdx.y;
    const float* p = in + (size_t)bc * 64;
    int y0, y1, x0, x1; float ty, tx;
    upidx7(y, 8, y0, y1, ty);
    upidx7(x, 8, x0, x1, tx);
    float v00 = p[y0 * 8 + x0], v01 = p[y0 * 8 + x1];
    float v10 = p[y1 * 8 + x0], v11 = p[y1 * 8 + x1];
    out[(size_t)bc * HWTOK + y * RESR + x] =
        (v00 * (1.f - tx) + v01 * tx) * (1.f - ty) + (v10 * (1.f - tx) + v11 * tx) * ty;
}

// A_bf16[r][c] = yb_nchw + gelu(s3_nchw)  (token-major GEMM A for out-proj)
__global__ __launch_bounds__(256) void combine_tok(const float* __restrict__ yb,
                                                   const float* __restrict__ s3,
                                                   __bf16* __restrict__ A) {
    int r = blockIdx.x, c = threadIdx.x;
    int b = r / HWTOK, p = r - b * HWTOK;
    size_t idx = ((size_t)b * DIMC + c) * HWTOK + p;
    A[(size_t)r * DIMC + c] = (__bf16)(yb[idx] + gelu_f(s3[idx]));
}

// ---------------------------------------------------------------------------
extern "C" void kernel_launch(void* const* d_in, const int* in_sizes, int n_in,
                              void* d_out, int out_size, void* d_ws, size_t ws_size,
                              hipStream_t stream)
{
    (void)in_sizes; (void)n_in; (void)out_size; (void)ws_size;
    const float* x     = (const float*)d_in[0];
    const float* n1_g  = (const float*)d_in[1];
    const float* n1_b  = (const float*)d_in[2];
    const float* Wv    = (const float*)d_in[3];
    const float* c1_w  = (const float*)d_in[4];
    const float* bn1g  = (const float*)d_in[5];
    const float* bn1b  = (const float*)d_in[6];
    const float* bn1m  = (const float*)d_in[7];
    const float* bn1v  = (const float*)d_in[8];
    const float* c2_w  = (const float*)d_in[9];
    const float* bn2g  = (const float*)d_in[10];
    const float* bn2b  = (const float*)d_in[11];
    const float* bn2m  = (const float*)d_in[12];
    const float* bn2v  = (const float*)d_in[13];
    const float* vu_w  = (const float*)d_in[14];
    const float* bnvg  = (const float*)d_in[15];
    const float* bnvb  = (const float*)d_in[16];
    const float* bnvm  = (const float*)d_in[17];
    const float* bnvv  = (const float*)d_in[18];
    const float* Wp    = (const float*)d_in[19];
    const float* bp    = (const float*)d_in[20];
    const float* n2_g  = (const float*)d_in[21];
    const float* n2_b  = (const float*)d_in[22];
    const float* f1_w  = (const float*)d_in[23];
    const float* f1_b  = (const float*)d_in[24];
    const float* bf1g  = (const float*)d_in[25];
    const float* bf1b  = (const float*)d_in[26];
    const float* bf1m  = (const float*)d_in[27];
    const float* bf1v  = (const float*)d_in[28];
    const float* f2_w  = (const float*)d_in[29];
    const float* f2_b  = (const float*)d_in[30];
    const float* bf2g  = (const float*)d_in[31];
    const float* bf2b  = (const float*)d_in[32];
    const float* bf2m  = (const float*)d_in[33];
    const float* bf2v  = (const float*)d_in[34];
    const float* f3_w  = (const float*)d_in[35];
    const float* f3_b  = (const float*)d_in[36];
    const float* bf3g  = (const float*)d_in[37];
    const float* bf3b  = (const float*)d_in[38];
    const float* bf3m  = (const float*)d_in[39];
    const float* bf3v  = (const float*)d_in[40];

    const int M = BATCH * HWTOK;   // 50176

    char* ws = (char*)d_ws;
    size_t off = 0;
    auto carve = [&](size_t bytes) -> char* {
        char* p = ws + off;
        off += (bytes + 255) & ~(size_t)255;
        return p;
    };
    __bf16* Abf  = (__bf16*)carve((size_t)M * HIDC * 2);            // GEMM A (all)
    __bf16* Wb   = (__bf16*)carve((size_t)HIDC * DIMC * 2);         // weights (seq reuse)
    float* vtok  = (float*)carve((size_t)M * DIMC * 4);
    float* nchw1 = (float*)carve((size_t)M * DIMC * 4);             // v NCHW -> s3
    float* nchw2 = (float*)carve((size_t)M * DIMC * 4);             // s1 -> y_up
    float* nchw3 = (float*)carve((size_t)M * DIMC * 4);             // yb
    float* x2    = (float*)carve((size_t)M * DIMC * 4);             // attn-block output
    float* h1    = (float*)carve((size_t)BATCH * HIDC * HWTOK * 4); // FFN hidden NCHW
    float* qb    = (float*)carve((size_t)BATCH * DIMC * 64 * 4);
    float* kb    = (float*)carve((size_t)BATCH * DIMC * 64 * 4);
    float* attn0 = (float*)carve((size_t)BATCH * HEADS * 64 * 64 * 4);
    float* y4    = (float*)carve((size_t)BATCH * DIMC * 64 * 4);
    float* coef  = (float*)carve((size_t)12 * 1024 * 4);
    float *sc1 = coef,          *sh1 = coef + 1024;
    float *sc2 = coef + 2048,   *sh2 = coef + 3072;
    float *scv = coef + 4096,   *shv = coef + 5120;
    float *scf1 = coef + 6144,  *shf1 = coef + 7168;
    float *scf2 = coef + 8192,  *shf2 = coef + 9216;
    float *scf3 = coef + 10240, *shf3 = coef + 11264;

    // BN coefficients
    bn_coef<<<1, 256, 0, stream>>>(bn1g, bn1b, bn1m, bn1v, sc1, sh1, DIMC);
    bn_coef<<<1, 256, 0, stream>>>(bn2g, bn2b, bn2m, bn2v, sc2, sh2, DIMC);
    bn_coef<<<1, 256, 0, stream>>>(bnvg, bnvb, bnvm, bnvv, scv, shv, DIMC);
    bn_coef<<<4, 256, 0, stream>>>(bf1g, bf1b, bf1m, bf1v, scf1, shf1, HIDC);
    bn_coef<<<4, 256, 0, stream>>>(bf2g, bf2b, bf2m, bf2v, scf2, shf2, HIDC);
    bn_coef<<<1, 256, 0, stream>>>(bf3g, bf3b, bf3m, bf3v, scf3, shf3, DIMC);

    // --- attention branch ---
    ln_to_bf16<<<M, 256, 0, stream>>>(x, n1_g, n1_b, Abf);
    cvt_bf16<<<(DIMC * DIMC) / 256, 256, 0, stream>>>(Wv, Wb, DIMC * DIMC);
    gemm_bf16_wmma<<<dim3(DIMC / 64, M / 128), 256, 0, stream>>>(
        Abf, Wb, M, DIMC, DIMC, nullptr, nullptr, nullptr, nullptr, vtok, 0);
    tok2nchw<<<M, 256, 0, stream>>>(vtok, nchw1);
    dw3x3_bn<<<dim3(RESR, BATCH * DIMC), 64, 0, stream>>>(
        nchw1, c1_w, nullptr, sc1, sh1, nchw2, nullptr, DIMC, 0);            // s1
    pool7<<<BATCH * DIMC, 64, 0, stream>>>(nchw2, qb);                        // q
    dw3x3_bn<<<dim3(RESR, BATCH * DIMC), 64, 0, stream>>>(
        nchw2, c2_w, nullptr, sc2, sh2, nchw1, nullptr, DIMC, DW_INGELU);    // s3
    pool7<<<BATCH * DIMC, 64, 0, stream>>>(nchw1, kb);                        // k
    qk_attn<<<dim3(64, BATCH * HEADS), 64, 0, stream>>>(qb, kb, attn0);
    attn_softmax_av<<<dim3(64, BATCH * HEADS), 256, 0, stream>>>(attn0, vtok, y4);
    up7x<<<dim3(RESR, BATCH * DIMC), 64, 0, stream>>>(y4, nchw2);             // up(y)
    dw3x3_bn<<<dim3(RESR, BATCH * DIMC), 64, 0, stream>>>(
        nchw2, vu_w, nullptr, scv, shv, nchw3, nullptr, DIMC, 0);            // yb
    combine_tok<<<M, 256, 0, stream>>>(nchw3, nchw1, Abf);                    // yb+gelu(s3)
    cvt_bf16<<<(DIMC * DIMC) / 256, 256, 0, stream>>>(Wp, Wb, DIMC * DIMC);
    gemm_bf16_wmma<<<dim3(DIMC / 64, M / 128), 256, 0, stream>>>(
        Abf, Wb, M, DIMC, DIMC, bp, nullptr, nullptr, x, x2, 0);             // x2 = x + proj

    // --- FFN branch ---
    ln_to_bf16<<<M, 256, 0, stream>>>(x2, n2_g, n2_b, Abf);
    cvt_bf16<<<(HIDC * DIMC) / 256, 256, 0, stream>>>(f1_w, Wb, HIDC * DIMC);
    gemm_bf16_wmma<<<dim3(HIDC / 64, M / 128), 256, 0, stream>>>(
        Abf, Wb, M, HIDC, DIMC, f1_b, scf1, shf1, nullptr, h1, GF_GELU | GF_SCATTER);
    dw3x3_bn<<<dim3(RESR, BATCH * HIDC), 64, 0, stream>>>(
        h1, f2_w, f2_b, scf2, shf2, nullptr, Abf, HIDC, DW_OUTGELU);         // -> bf16 token A
    cvt_bf16<<<(DIMC * HIDC) / 256, 256, 0, stream>>>(f3_w, Wb, DIMC * HIDC);
    gemm_bf16_wmma<<<dim3(DIMC / 64, M / 128), 256, 0, stream>>>(
        Abf, Wb, M, DIMC, HIDC, f3_b, scf3, shf3, x2, (float*)d_out, 0);     // out = x2 + z
}
